// equivariant_decoder_5566277616513
// MI455X (gfx1250) — matrix operand order repaired and checked by reading, not compiled, not run on hardware
//
#include <hip/hip_runtime.h>
#include <math.h>

typedef __bf16 v8bf  __attribute__((ext_vector_type(8)));
typedef __bf16 v16bf __attribute__((ext_vector_type(16)));
typedef float  v8f   __attribute__((ext_vector_type(8)));
typedef unsigned int v4u __attribute__((ext_vector_type(4)));
typedef int v8i __attribute__((ext_vector_type(8)));
typedef int v4i __attribute__((ext_vector_type(4)));

#define BATCH 4096
#define DEPTH 768
#define NIR   256
#define PDEC  256
#define OUTC  3328           // P*3 + P*(NUM_CLASSES+1) = 768 + 2560
#define EPSLN 1e-5f

#if __has_builtin(__builtin_amdgcn_tensor_load_to_lds)
#define USE_TDM 1
#endif

// ---------------- workspace layout (bytes) ----------------
static const size_t OFF_WTP   = 0;                                        // w_tp as [j][u][i] bf16: 33,554,432
static const size_t OFF_WTIN  = OFF_WTP   + (size_t)256*256*256*2;        // W_in^T  [768][256] bf16
static const size_t OFF_WTH   = OFF_WTIN  + (size_t)768*256*2;            // Wh^T    [2][768][768] bf16
static const size_t OFF_WTOUT = OFF_WTH   + (size_t)2*768*768*2;          // W_out^T [256][768] bf16
static const size_t OFF_RBF   = OFF_WTOUT + (size_t)256*768*2;            // r bf16  [4096][256]
static const size_t OFF_H     = OFF_RBF   + (size_t)4096*256*2;           // h f32   [4096][768]
static const size_t OFF_LN    = OFF_H     + (size_t)4096*768*4;           // ln bf16 [4096][768] (reused as h_bf16)
static const size_t OFF_RDEC  = OFF_LN    + (size_t)4096*768*2;           // r_dec bf16 [4096][256]
// total ~59.8 MB

// ---------------- fragment helpers ----------------
__device__ __forceinline__ v16bf frag_cat(v8bf lo, v8bf hi) {
  return __builtin_shufflevector(lo, hi, 0,1,2,3,4,5,6,7,8,9,10,11,12,13,14,15);
}
// A-matrix (16x32 bf16) per-lane layout: contiguous K chunk of 16 at half*16
__device__ __forceinline__ v16bf ld_afrag(const __bf16* p) {
  v8bf lo = *(const v8bf*)p;
  v8bf hi = *(const v8bf*)(p + 8);
  return frag_cat(lo, hi);
}
// B-matrix (32x16 bf16) per-lane layout: contiguous K chunk of 16 at half*16
__device__ __forceinline__ v16bf ld_bfrag(const __bf16* p) {
  v8bf lo = *(const v8bf*)p;
  v8bf hi = *(const v8bf*)(p + 8);
  return frag_cat(lo, hi);
}
__device__ __forceinline__ float gelu_tanh(float x) {
  float x3 = x * x * x;
  return 0.5f * x * (1.0f + tanhf(0.7978845608028654f * (x + 0.044715f * x3)));
}

#ifdef USE_TDM
// Issue a 1-row TDM DMA: 4096 bf16 elements (8KB) global -> LDS.
// D# layout per CDNA5 ISA ch.8: group0 {count, lds_addr, global_addr, type=2},
// group1 {data_size=2B, tensor_dim0=4096, tensor_dim1=1, tile_dim0=4096,
//         tensor_dim0_stride=4096}. Groups 2/3 unused (<=2D tensor).
__device__ __forceinline__ void tdm_load_tile(const __bf16* gsrc, void* ldst) {
  unsigned long long ga = (unsigned long long)(uintptr_t)gsrc;
  unsigned int laddr = (unsigned int)(uintptr_t)ldst;   // LDS byte offset (addr[31:0])
  v4u g0;
  g0.x = 1u;                                                 // count=1, user mode
  g0.y = laddr;                                              // lds_addr [63:32]
  g0.z = (unsigned int)(ga & 0xFFFFFFFFu);                   // global_addr lo
  g0.w = (unsigned int)((ga >> 32) & 0x1FFFFFFu) | (2u << 30); // addr[56:32] | type=2
  v8i g1;
  g1[0] = 0x00010000;                 // wg_mask=0, data_size=1 (2 bytes)
  g1[1] = (int)(4096u << 16);         // tensor_dim0[15:0]=4096 at bits[63:48]
  g1[2] = (int)(1u << 16);            // tensor_dim0[31:16]=0, tensor_dim1[15:0]=1
  g1[3] = (int)(4096u << 16);         // tensor_dim1[31:16]=0, tile_dim0=4096
  g1[4] = 0;                          // tile_dim1=0, tile_dim2=0 (unused)
  g1[5] = 4096;                       // tensor_dim0_stride lo = 4096
  g1[6] = 0;                          // stride0 hi, stride1 lo
  g1[7] = 0;                          // stride1 hi
  v4i z4 = {0, 0, 0, 0};
  v8i z8 = {0, 0, 0, 0, 0, 0, 0, 0};
  // 6-arg form (clang-23 / therock-10.0 lane): groups 0..3 + extra group + cpol
  __builtin_amdgcn_tensor_load_to_lds(g0, g1, z4, z4, z8, 0);
}
#endif

// ---------------- utility kernels ----------------
__global__ void k_zero(float* __restrict__ p, int n) {
  int i = blockIdx.x * blockDim.x + threadIdx.x;
  if (i < n) p[i] = 0.0f;
}

__global__ void k_f32_to_bf16(const float* __restrict__ in, __bf16* __restrict__ out, int n) {
  int i = blockIdx.x * blockDim.x + threadIdx.x;
  if (i < n) out[i] = (__bf16)in[i];
}

// tiled transpose + convert: in [R x C] f32  ->  out [C x R] bf16
__global__ void k_transpose_cvt(const float* __restrict__ in, __bf16* __restrict__ out,
                                int R, int C) {
  __shared__ float tile[32][33];
  int c0 = blockIdx.x * 32, r0 = blockIdx.y * 32;
  int tx = threadIdx.x, ty = threadIdx.y;
#pragma unroll
  for (int i = 0; i < 4; ++i) {
    int r = r0 + ty + i * 8;
    tile[ty + i * 8][tx] = in[(size_t)r * C + c0 + tx];
  }
  __syncthreads();
#pragma unroll
  for (int i = 0; i < 4; ++i) {
    int c = c0 + ty + i * 8;
    out[(size_t)c * R + r0 + tx] = (__bf16)tile[tx][ty + i * 8];
  }
}

// r[b,n] = ||enc[b,n,:]||  -> bf16
__global__ void k_rnorm(const float* __restrict__ enc, __bf16* __restrict__ rbf) {
  int idx = blockIdx.x * blockDim.x + threadIdx.x;   // 4096*256 threads
  int b = idx >> 8, n = idx & 255;
  const float* e = enc + (size_t)b * DEPTH + n * 3;
  float x = e[0], y = e[1], z = e[2];
  rbf[idx] = (__bf16)sqrtf(x * x + y * y + z * z);
}

// LayerNorm over 768 cols, scaled by gamma/beta, emits bf16
__global__ void k_layernorm(const float* __restrict__ h, const float* __restrict__ g,
                            const float* __restrict__ bta, __bf16* __restrict__ lnout) {
  __shared__ float ssum[256], ssq[256], stat[2];
  int b = blockIdx.x, t = threadIdx.x;
  const float* hr = h + (size_t)b * DEPTH;
  float x0 = hr[t], x1 = hr[t + 256], x2 = hr[t + 512];
  ssum[t] = x0 + x1 + x2;
  ssq[t]  = x0 * x0 + x1 * x1 + x2 * x2;
  __syncthreads();
  for (int off = 128; off > 0; off >>= 1) {
    if (t < off) { ssum[t] += ssum[t + off]; ssq[t] += ssq[t + off]; }
    __syncthreads();
  }
  if (t == 0) {
    float mu = ssum[0] * (1.0f / 768.0f);
    float var = ssq[0] * (1.0f / 768.0f) - mu * mu;
    stat[0] = mu; stat[1] = rsqrtf(var + EPSLN);
  }
  __syncthreads();
  float mu = stat[0], rs = stat[1];
  __bf16* o = lnout + (size_t)b * DEPTH;
  o[t]       = (__bf16)((x0 - mu) * rs * g[t]       + bta[t]);
  o[t + 256] = (__bf16)((x1 - mu) * rs * g[t + 256] + bta[t + 256]);
  o[t + 512] = (__bf16)((x2 - mu) * rs * g[t + 512] + bta[t + 512]);
}

// ---------------- WMMA GEMM for the MLP ----------------
// C[4096 x NN] = A[4096 x K](bf16, row-major K) @ Bt[NN x K]^T (bf16) + bias
// MODE 0: H = val + bias            (in-projection)
// MODE 1: H += gelu(val + bias)     (residual hidden block)
// MODE 2: Obf = bf16(val + bias)    (out-projection)
template <int MODE, int K, int NN>
__global__ __launch_bounds__(256) void k_gemm(const __bf16* __restrict__ A,
                                              const __bf16* __restrict__ Bt,
                                              const float* __restrict__ bias,
                                              float* __restrict__ H,
                                              __bf16* __restrict__ Obf) {
  int lane = threadIdx.x & 31, wave = threadIdx.x >> 5;
  int half = lane >> 4, l16 = lane & 15;
  int m0 = blockIdx.y * 128 + (wave >> 1) * 32;
  int n0 = blockIdx.x * 64  + (wave & 1) * 32;
  v8f c[2][2] = {};
  const __bf16* a0 = A  + (size_t)(m0 + l16) * K;
  const __bf16* a1 = A  + (size_t)(m0 + 16 + l16) * K;
  const __bf16* b0 = Bt + (size_t)(n0 + l16) * K;
  const __bf16* b1 = Bt + (size_t)(n0 + 16 + l16) * K;
#pragma unroll 4
  for (int k = 0; k < K; k += 32) {
    v16bf fa0 = ld_afrag(a0 + k + half * 16);
    v16bf fa1 = ld_afrag(a1 + k + half * 16);
    v16bf fb0 = ld_bfrag(b0 + k + half * 16);
    v16bf fb1 = ld_bfrag(b1 + k + half * 16);
    c[0][0] = __builtin_amdgcn_wmma_f32_16x16x32_bf16(false, fa0, false, fb0, (short)0, c[0][0], false, false);
    c[0][1] = __builtin_amdgcn_wmma_f32_16x16x32_bf16(false, fa0, false, fb1, (short)0, c[0][1], false, false);
    c[1][0] = __builtin_amdgcn_wmma_f32_16x16x32_bf16(false, fa1, false, fb0, (short)0, c[1][0], false, false);
    c[1][1] = __builtin_amdgcn_wmma_f32_16x16x32_bf16(false, fa1, false, fb1, (short)0, c[1][1], false, false);
  }
#pragma unroll
  for (int mt = 0; mt < 2; ++mt)
#pragma unroll
    for (int nt = 0; nt < 2; ++nt)
#pragma unroll
      for (int v = 0; v < 8; ++v) {
        int row = m0 + mt * 16 + half * 8 + v;
        int col = n0 + nt * 16 + l16;
        float val = c[mt][nt][v] + bias[col];
        if constexpr (MODE == 0) {
          H[(size_t)row * NN + col] = val;
        } else if constexpr (MODE == 1) {
          size_t o = (size_t)row * NN + col;
          H[o] += gelu_tanh(val);
        } else {
          Obf[(size_t)row * NN + col] = (__bf16)val;
        }
      }
}

// ---------------- fused tensor product ----------------
// Per workgroup: 128 batches x 16 u's. For each j (256): compute
// A_tile[b(16),u(16)] = sum_i rdec[b,i]*wtp[j][u][i] via 8 WMMAs per wave,
// then vec[b,u,m] += A_tile * enc[b,j,m]. B tiles double-buffered in LDS
// (staged by TDM tensor_load_to_lds when available), shared across the 8
// waves; enc staged per 16-j chunk in LDS.
__global__ __launch_bounds__(256) void k_tensorproduct(const float* __restrict__ enc,
                                                       const __bf16* __restrict__ rdec,
                                                       const __bf16* __restrict__ wtp,
                                                       float* __restrict__ out) {
  __shared__ __bf16 lds_b[2][16 * 256];   // [u_local][i], 8KB each
  __shared__ float  lds_enc[128 * 48];    // [b_local][16 j * 3 m]
  int t = threadIdx.x;
  int lane = t & 31, wave = t >> 5;
  int half = lane >> 4, l16 = lane & 15;
  int wg = blockIdx.x;
  int b_base = (wg >> 4) * 128;
  int u0 = (wg & 15) << 4;
  int b0 = b_base + wave * 16;

  // A fragments: rdec rows b0..b0+15, all 256 i (constant over j loop)
  v16bf af[8];
  const __bf16* arow = rdec + (size_t)(b0 + l16) * 256;
#pragma unroll
  for (int s = 0; s < 8; ++s) af[s] = ld_afrag(arow + s * 32 + half * 16);

  float vacc[8][3] = {};

  for (int jc = 0; jc < 256; jc += 16) {
    __syncthreads();   // protect lds_enc and lds_b reuse across chunks
    {  // stage enc[b_base..+127, jc..jc+15, 0..2] -> lds_enc
      int bl = t >> 1, part = t & 1;
      const float4* src = (const float4*)(enc + (size_t)(b_base + bl) * DEPTH + jc * 3 + part * 24);
      float4* dst = (float4*)(lds_enc + bl * 48 + part * 24);
#pragma unroll
      for (int q = 0; q < 6; ++q) dst[q] = src[q];
    }
    // stage B tile for j = jc into buffer 0 (16u x 256i contiguous 8KB)
#ifdef USE_TDM
    if (wave == 0) {
      tdm_load_tile(wtp + (((size_t)jc * 256 + u0) << 8), &lds_b[0][0]);
      __builtin_amdgcn_s_wait_tensorcnt(0);
    }
#else
    {
      const v8bf* src = (const v8bf*)(wtp + (((size_t)jc * 256 + u0) << 8)) + t * 2;
      v8bf* dst = (v8bf*)lds_b[0] + t * 2;
      dst[0] = src[0]; dst[1] = src[1];
    }
#endif
    __syncthreads();
    for (int jj = 0; jj < 16; ++jj) {
      int cur = jj & 1;
      if (jj < 15) {  // stage next-j B tile into the other buffer (overlaps compute)
        int jn = jc + jj + 1;
        const __bf16* sp = wtp + (((size_t)jn * 256 + u0) << 8);
#ifdef USE_TDM
        if (wave == 0) tdm_load_tile(sp, &lds_b[cur ^ 1][0]);
#else
        const v8bf* src = (const v8bf*)sp + t * 2;
        v8bf* dst = (v8bf*)lds_b[cur ^ 1] + t * 2;
        v8bf s0 = src[0], s1 = src[1];
        dst[0] = s0; dst[1] = s1;
#endif
        if (jn + 1 < 256)  // L2 prefetch hint for j+2 (global_prefetch_b8)
          __builtin_prefetch(wtp + (((size_t)(jn + 1) * 256 + u0) << 8) + (t & 31) * 256, 0, 1);
      }
      // i-contraction: A_tile[b,u] for this j
      v8f c = {};
      const __bf16* bb = &lds_b[cur][l16 * 256 + half * 16];
#pragma unroll
      for (int s = 0; s < 8; ++s) {
        v16bf fb = ld_bfrag(bb + s * 32);
        c = __builtin_amdgcn_wmma_f32_16x16x32_bf16(false, af[s], false, fb, (short)0, c, false, false);
      }
      // fold into vec accumulators against enc[b, j, m]
      const float* ep = lds_enc + (wave * 16 + half * 8) * 48 + jj * 3;
#pragma unroll
      for (int v = 0; v < 8; ++v) {
        float a = c[v];
        const float* e = ep + v * 48;
        vacc[v][0] += a * e[0];
        vacc[v][1] += a * e[1];
        vacc[v][2] += a * e[2];
      }
#ifdef USE_TDM
      if (jj < 15 && wave == 0) __builtin_amdgcn_s_wait_tensorcnt(0);
#endif
      __syncthreads();
    }
  }
  // write vec block: out[b, u*3+m] * (1/N)
  const float alpha = 1.0f / 256.0f;
  int u = u0 + l16;
#pragma unroll
  for (int v = 0; v < 8; ++v) {
    int row = b0 + half * 8 + v;
    float* o = out + (size_t)row * OUTC + u * 3;
    o[0] = vacc[v][0] * alpha;
    o[1] = vacc[v][1] * alpha;
    o[2] = vacc[v][2] * alpha;
  }
}

// ---------------- host launcher ----------------
extern "C" void kernel_launch(void* const* d_in, const int* in_sizes, int n_in,
                              void* d_out, int out_size, void* d_ws, size_t ws_size,
                              hipStream_t stream) {
  (void)in_sizes; (void)n_in; (void)out_size; (void)ws_size;
  const float* enc   = (const float*)d_in[0];
  const float* W_in  = (const float*)d_in[2];
  const float* b_in  = (const float*)d_in[3];
  const float* Wh    = (const float*)d_in[4];
  const float* bh    = (const float*)d_in[5];
  const float* gamma = (const float*)d_in[6];
  const float* beta  = (const float*)d_in[7];
  const float* W_out = (const float*)d_in[8];
  const float* b_out = (const float*)d_in[9];
  const float* w_tp  = (const float*)d_in[10];
  float* out = (float*)d_out;
  char* ws = (char*)d_ws;

  __bf16* wtpbf = (__bf16*)(ws + OFF_WTP);
  __bf16* wtin  = (__bf16*)(ws + OFF_WTIN);
  __bf16* wth   = (__bf16*)(ws + OFF_WTH);
  __bf16* wtout = (__bf16*)(ws + OFF_WTOUT);
  __bf16* rbf   = (__bf16*)(ws + OFF_RBF);
  float*  h     = (float*) (ws + OFF_H);
  __bf16* lnbf  = (__bf16*)(ws + OFF_LN);    // also reused as h_bf16
  __bf16* rdec  = (__bf16*)(ws + OFF_RDEC);

  // 0) zero the full output (scalar 0e block stays zero; vec block rewritten)
  {
    int n = BATCH * OUTC;
    k_zero<<<(n + 255) / 256, 256, 0, stream>>>(out, n);
  }
  // 1) weight conversions/transposes (w_tp [i][j][u] -> [j][u][i], W -> W^T), f32 -> bf16
  dim3 tb(32, 8);
  k_transpose_cvt<<<dim3(65536 / 32, 256 / 32), tb, 0, stream>>>(w_tp, wtpbf, 256, 65536);
  k_transpose_cvt<<<dim3(768 / 32, 256 / 32), tb, 0, stream>>>(W_in, wtin, 256, 768);
  k_transpose_cvt<<<dim3(768 / 32, 768 / 32), tb, 0, stream>>>(Wh, wth, 768, 768);
  k_transpose_cvt<<<dim3(768 / 32, 768 / 32), tb, 0, stream>>>(Wh + 768 * 768, wth + 768 * 768, 768, 768);
  k_transpose_cvt<<<dim3(256 / 32, 768 / 32), tb, 0, stream>>>(W_out, wtout, 768, 256);
  // 2) vector norms
  k_rnorm<<<(BATCH * NIR) / 256, 256, 0, stream>>>(enc, rbf);
  // 3) MLP: in-projection
  k_gemm<0, 256, 768><<<dim3(12, 32), 256, 0, stream>>>(rbf, wtin, b_in, h, nullptr);
  // 4) residual hidden blocks
  for (int l = 0; l < 2; ++l) {
    k_layernorm<<<BATCH, 256, 0, stream>>>(h, gamma + l * 768, beta + l * 768, lnbf);
    k_gemm<1, 768, 768><<<dim3(12, 32), 256, 0, stream>>>(lnbf, wth + (size_t)l * 768 * 768,
                                                          bh + l * 768, h, nullptr);
  }
  // 5) out-projection (h -> bf16 first; reuse LN buffer)
  k_f32_to_bf16<<<(BATCH * DEPTH) / 256, 256, 0, stream>>>(h, lnbf, BATCH * DEPTH);
  k_gemm<2, 768, 256><<<dim3(4, 32), 256, 0, stream>>>(lnbf, wtout, b_out, nullptr, rdec);
  // 6) fused equivariant tensor product (WMMA over i, TDM-staged B tiles)
  k_tensorproduct<<<(BATCH / 128) * (PDEC / 16), 256, 0, stream>>>(enc, rdec, wtpbf, out);
}
